// M_12283606467103
// MI455X (gfx1250) — compile-verified, matrix operand order checked by reading, not run on hardware
//
#include <hip/hip_runtime.h>

// Attention epilogue: softmax(scores) -> dropout(mask, keep=0.9) -> P @ V
// B=8, H=12, S=1024, D=64.  HBM-bound (~554 MB @ 23.3 TB/s ~= 24us floor):
// scores read once from HBM; softmax + f16 P staged in LDS; P@V done with
// v_wmma_f32_16x16x32_f16 (fp32 accumulate); 1/(rowsum*keep) folded into the
// output scale.

typedef __attribute__((ext_vector_type(4)))  _Float16 v4h;
typedef __attribute__((ext_vector_type(8)))  _Float16 v8h;
typedef __attribute__((ext_vector_type(16))) _Float16 v16h;
typedef __attribute__((ext_vector_type(8)))  float    v8f;

#define S_LEN    1024
#define D_DIM    64
#define MBLK     64          // query rows per workgroup
#define NTHREADS 512         // 16 wave32
#define KPAD     (S_LEN + 8) // half-elements per LDS row; 516 dwords -> bank-conflict free
#define KEEP     0.9f

__global__ __launch_bounds__(NTHREADS) void softmax_dropout_pv(
    const float* __restrict__ scores,
    const float* __restrict__ v,
    const unsigned char* __restrict__ mask,  // JAX bool -> 1 byte/elem
    float* __restrict__ out)
{
  extern __shared__ char smem[];
  _Float16* Vt  = (_Float16*)smem;                      // [D_DIM][KPAD]  V^T as f16
  _Float16* P   = Vt + (size_t)D_DIM * KPAD;            // [MBLK][KPAD]   masked exp(s-max) as f16
  float*    red = (float*)(P + (size_t)MBLK * KPAD);    // [MBLK][8] partials
  float*    rmax = red + MBLK * 8;                      // [MBLK]
  float*    rinv = rmax + MBLK;                         // [MBLK]  (1/keep)/rowsum

  const int bh = blockIdx.y;           // 0..B*H-1
  const int q0 = blockIdx.x * MBLK;    // query-row base
  const int t  = threadIdx.x;

  const size_t scoreBase = ((size_t)bh * S_LEN + q0) * S_LEN;
  const size_t vBase     = (size_t)bh * S_LEN * D_DIM;

  // ---- Phase 0: stage V [S][D] fp32 -> Vt [D][KPAD] f16 (transposed) ----
  {
    const float4* v4 = (const float4*)(v + vBase);
    #pragma unroll
    for (int p = 0; p < (S_LEN * D_DIM / 4) / NTHREADS; ++p) {
      int f  = t + p * NTHREADS;       // float4 index; coalesced across lanes
      int k  = f >> 4;                 // D/4 = 16 float4 per k-row
      int n0 = (f & 15) * 4;
      float4 val = v4[f];
      Vt[(size_t)(n0 + 0) * KPAD + k] = (_Float16)val.x;
      Vt[(size_t)(n0 + 1) * KPAD + k] = (_Float16)val.y;
      Vt[(size_t)(n0 + 2) * KPAD + k] = (_Float16)val.z;
      Vt[(size_t)(n0 + 3) * KPAD + k] = (_Float16)val.w;
    }
  }

  // ---- Phase 1a: row max (8 threads per row, float4 streaming) ----
  const int r = t >> 3;                // 0..63 : query row within block
  const int c = t & 7;                 // 0..7  : column-group within row
  const float4* srow = (const float4*)(scores + scoreBase + (size_t)r * S_LEN);
  float mx = -3.4e38f;
  #pragma unroll 4
  for (int j = 0; j < 32; ++j) {
    float4 s4 = srow[c + 8 * j];       // 8 lanes x 16B = 128B coalesced
    mx = fmaxf(mx, fmaxf(fmaxf(s4.x, s4.y), fmaxf(s4.z, s4.w)));
  }
  red[r * 8 + c] = mx;
  __syncthreads();
  if (t < MBLK) {
    float m = red[t * 8];
    #pragma unroll
    for (int i = 1; i < 8; ++i) m = fmaxf(m, red[t * 8 + i]);
    rmax[t] = m;
  }
  __syncthreads();

  // ---- Phase 1b: e = exp(s-max); rowsum over ALL e; P = mask ? e : 0 (f16) ----
  const float rm = rmax[r];
  const unsigned char* mrow = mask + scoreBase + (size_t)r * S_LEN;
  _Float16* prow = P + (size_t)r * KPAD;
  float sum = 0.f;
  #pragma unroll 4
  for (int j = 0; j < 32; ++j) {
    int fi = c + 8 * j;
    int k  = fi * 4;
    float4 s4 = srow[fi];                              // L2 hit (2nd pass, block=256KB)
    unsigned int mw = *(const unsigned int*)(mrow + k);
    float e0 = __expf(s4.x - rm);
    float e1 = __expf(s4.y - rm);
    float e2 = __expf(s4.z - rm);
    float e3 = __expf(s4.w - rm);
    sum += (e0 + e1) + (e2 + e3);                      // denominator: pre-dropout
    v4h pv;
    pv[0] = (mw & 0x000000ffu) ? (_Float16)e0 : (_Float16)0.f;
    pv[1] = (mw & 0x0000ff00u) ? (_Float16)e1 : (_Float16)0.f;
    pv[2] = (mw & 0x00ff0000u) ? (_Float16)e2 : (_Float16)0.f;
    pv[3] = (mw & 0xff000000u) ? (_Float16)e3 : (_Float16)0.f;
    *(v4h*)(prow + k) = pv;
  }
  red[r * 8 + c] = sum;
  __syncthreads();
  if (t < MBLK) {
    float s = 0.f;
    #pragma unroll
    for (int i = 0; i < 8; ++i) s += red[t * 8 + i];
    rinv[t] = (1.0f / KEEP) / s;
  }
  __syncthreads();

  // ---- Phase 2: per-wave 16x16 output tile, K=1024 in steps of 32 ----
  const int w    = t >> 5;             // wave 0..15
  const int lane = t & 31;
  const int mt   = w >> 2;             // M tile 0..3
  const int nt   = w & 3;              // N tile 0..3
  const int lrow = lane & 15;
  const int kb   = (lane >> 4) * 8;    // 16-bit A/B layout: hi half-wave K-offset 8

  const _Float16* arow = P  + (size_t)(mt * 16 + lrow) * KPAD;  // A: row M=lrow
  const _Float16* brow = Vt + (size_t)(nt * 16 + lrow) * KPAD;  // B: col N=lrow (Vt is V^T)
  v8f acc = {};
  #pragma unroll 4
  for (int k = 0; k < S_LEN; k += 32) {
    v8h alo = *(const v8h*)(arow + k + kb);        // K = k+kb   .. +7  (b128)
    v8h ahi = *(const v8h*)(arow + k + 16 + kb);   // K = k+16+kb.. +7  (b128)
    v8h blo = *(const v8h*)(brow + k + kb);
    v8h bhi = *(const v8h*)(brow + k + 16 + kb);
    v16h a = __builtin_shufflevector(alo, ahi, 0,1,2,3,4,5,6,7,8,9,10,11,12,13,14,15);
    v16h b = __builtin_shufflevector(blo, bhi, 0,1,2,3,4,5,6,7,8,9,10,11,12,13,14,15);
    acc = __builtin_amdgcn_wmma_f32_16x16x32_f16(
        /*neg_a=*/false, a, /*neg_b=*/false, b,
        /*c_mod=*/(short)0, acc, /*reuse_a=*/false, /*reuse_b=*/false);
  }

  // C layout: lanes 0-15 -> M=r, N=lane ; lanes 16-31 -> M=8+r, N=lane-16
  const int roff = (lane >> 4) * 8;
  float* orow = out + ((size_t)bh * S_LEN + q0 + mt * 16 + roff) * D_DIM
                    + nt * 16 + lrow;
  #pragma unroll
  for (int i = 0; i < 8; ++i) {
    float sc = rinv[mt * 16 + roff + i];
    orow[(size_t)i * D_DIM] = acc[i] * sc;
  }
}

extern "C" void kernel_launch(void* const* d_in, const int* in_sizes, int n_in,
                              void* d_out, int out_size, void* d_ws, size_t ws_size,
                              hipStream_t stream) {
  (void)in_sizes; (void)n_in; (void)out_size; (void)d_ws; (void)ws_size;
  const float*         scores = (const float*)d_in[0];
  const float*         v      = (const float*)d_in[1];
  const unsigned char* mask   = (const unsigned char*)d_in[2];
  float*               out    = (float*)d_out;

  const int B = 8, H = 12;
  dim3 grid(S_LEN / MBLK, B * H);   // 16 x 96 workgroups
  size_t smem = (size_t)(D_DIM + MBLK) * KPAD * sizeof(_Float16)   // Vt + P
              + (size_t)(MBLK * 8 + MBLK + MBLK) * sizeof(float);  // red + rmax + rinv
  softmax_dropout_pv<<<grid, NTHREADS, smem, stream>>>(scores, v, mask, out);
}
